// EulerFullAttention_75892072121053
// MI455X (gfx1250) — compile-verified, matrix operand order checked by reading, not run on hardware
//
#include <hip/hip_runtime.h>
#include <math.h>

typedef __attribute__((ext_vector_type(16))) _Float16 v16h;
typedef __attribute__((ext_vector_type(8)))  _Float16 v8h;
typedef __attribute__((ext_vector_type(8)))  float    v8f;

#define BD 2
#define SD 2048
#define HD 16
#define DH 64
#define QK 128   /* 2*dh: Euler features per head for Q/K */

// ---------------------------------------------------------------------------
// Kernel 1: build f16 Q (B,H,S,128), K (B,H,S,128) and V^T (B,H,64,S)
// ---------------------------------------------------------------------------
__global__ void euler_qkv_kernel(const float* __restrict__ x,
                                 const float* __restrict__ w_q, const float* __restrict__ b_q,
                                 const float* __restrict__ phi_q,
                                 const float* __restrict__ w_k, const float* __restrict__ b_k,
                                 const float* __restrict__ phi_k,
                                 const float* __restrict__ w_v, const float* __restrict__ b_v,
                                 _Float16* __restrict__ Qb, _Float16* __restrict__ Kb,
                                 _Float16* __restrict__ Vt)
{
    int idx = blockIdx.x * blockDim.x + threadIdx.x;     // flat (b,s,h,j)
    if (idx >= BD * SD * HD * DH) return;
    int j = idx & 63;
    int h = (idx >> 6) & 15;
    int s = (idx >> 10) & 2047;
    int b = idx >> 21;
    float xv = x[idx];
    int   hj = h * DH + j;
    float t  = (float)s;

    float thq = xv / (1.0f + fabsf(w_q[hj])) + b_q[hj] + t * phi_q[hj];
    float sq, cq; __sincosf(thq, &sq, &cq);
    float thk = xv / (1.0f + fabsf(w_k[hj])) + b_k[hj] + t * phi_k[hj];
    float sk, ck; __sincosf(thk, &sk, &ck);
    float thv = xv / (1.0f + fabsf(w_v[hj])) + b_v[hj];
    float sv, cv; __sincosf(thv, &sv, &cv);

    size_t bh = (size_t)(b * HD + h);
    size_t qo = (bh * SD + (size_t)s) * QK + j;
    Qb[qo]      = (_Float16)cq;
    Qb[qo + DH] = (_Float16)sq;
    Kb[qo]      = (_Float16)ck;
    Kb[qo + DH] = (_Float16)sk;
    Vt[(bh * DH + (size_t)j) * SD + s] = (_Float16)(cv + sv);
}

// Load one WMMA row fragment: halves [0..7] from row+klo, [8..15] from row+khi.
// Matches the 16-bit A/B fragment k-mapping: kmap(e) = e<8 ? grp*8+e : 16+grp*8+(e-8).
__device__ inline v16h load_frag_row(const _Float16* __restrict__ row, int klo, int khi)
{
    v8h lo = *(const v8h*)(row + klo);
    v8h hi = *(const v8h*)(row + khi);
    v16h a;
#pragma unroll
    for (int i = 0; i < 8; ++i) { a[i] = lo[i]; a[i + 8] = hi[i]; }
    return a;
}

// ---------------------------------------------------------------------------
// Kernel 2: causal flash attention, one wave32 per 16-query tile.
//   GEMM1: S^T = K_tile(16x32..) x Q^T           (v_wmma_f32_16x16x32_f16)
//   GEMM2: O^T += V^T_tile(16x32) x P^T          (D-layout of S^T == B-layout of P^T)
// Epilogue folds the final cos+sin output transform.
// ---------------------------------------------------------------------------
__global__ void __launch_bounds__(128)
euler_flash_attn_kernel(const _Float16* __restrict__ Qb, const _Float16* __restrict__ Kb,
                        const _Float16* __restrict__ Vt,
                        const float* __restrict__ w_out, const float* __restrict__ b_out,
                        float* __restrict__ out)
{
    const int lane = threadIdx.x & 31;
    const int wave = threadIdx.x >> 5;
    const int wid  = blockIdx.x * 4 + wave;      // global wave id
    const int qt = wid & 127;                    // S/16 = 128 query tiles
    const int h  = (wid >> 7) & 15;
    const int b  = wid >> 11;

    const int n   = lane & 15;                   // column index inside a 16-wide tile
    const int grp = lane >> 4;                   // half-lane group

    const int q0 = qt * 16;
    const int qg = q0 + n;                       // this lane's query row

    const size_t bh = (size_t)(b * HD + h);
    const _Float16* Qrow  = Qb + (bh * SD + (size_t)(q0 + n)) * QK;
    const _Float16* Kbase = Kb + bh * SD * QK;
    const _Float16* Vbase = Vt + bh * DH * SD;

    // Q^T B-fragments: 4 slices of 32 along the 128-wide feature dim, kept in VGPRs.
    v16h qf[4];
#pragma unroll
    for (int sd = 0; sd < 4; ++sd)
        qf[sd] = load_frag_row(Qrow, sd * 32 + grp * 8, sd * 32 + 16 + grp * 8);

    v8f acc[4];                                  // O^T: 4 feature tiles of 16
#pragma unroll
    for (int t = 0; t < 4; ++t) acc[t] = (v8f){};
    float m_run = -3.0e38f;
    float l_run = 0.0f;
    const float inv_scale = 0.08838834764831845f;   // 1/sqrt(2*dh)

    const int kb_end = (q0 + 15) >> 5;           // causal: keys <= q0+15
    for (int kb = 0; kb <= kb_end; ++kb) {
        const int kbase = kb * 32;

        // ---- GEMM1: two 16x16 S^T tiles over a 32-key block ----
        v8f s0 = (v8f){}, s1 = (v8f){};
        const _Float16* Krow0 = Kbase + (size_t)(kbase + n) * QK;
        const _Float16* Krow1 = Kbase + (size_t)(kbase + 16 + n) * QK;
#pragma unroll
        for (int sd = 0; sd < 4; ++sd) {
            v16h k0 = load_frag_row(Krow0, sd * 32 + grp * 8, sd * 32 + 16 + grp * 8);
            s0 = __builtin_amdgcn_wmma_f32_16x16x32_f16(false, k0, false, qf[sd],
                                                        (short)0, s0, false, false);
        }
#pragma unroll
        for (int sd = 0; sd < 4; ++sd) {
            v16h k1 = load_frag_row(Krow1, sd * 32 + grp * 8, sd * 32 + 16 + grp * 8);
            s1 = __builtin_amdgcn_wmma_f32_16x16x32_f16(false, k1, false, qf[sd],
                                                        (short)0, s1, false, false);
        }

        // ---- online softmax over keys (per query = lane&15, partner via xor-16) ----
        float sc[16];
#pragma unroll
        for (int r = 0; r < 8; ++r) {
            int kg0 = kbase + r + 8 * grp;
            int kg1 = kbase + 16 + r + 8 * grp;
            sc[r]     = (kg0 <= qg) ? s0[r] * inv_scale : -3.0e38f;
            sc[8 + r] = (kg1 <= qg) ? s1[r] * inv_scale : -3.0e38f;
        }
        float mloc = sc[0];
#pragma unroll
        for (int i = 1; i < 16; ++i) mloc = fmaxf(mloc, sc[i]);
        mloc = fmaxf(mloc, __shfl_xor(mloc, 16, 32));
        float m_new = fmaxf(m_run, mloc);
        float corr  = __expf(m_run - m_new);
        float psum  = 0.0f;
        v16h pf;                                 // P^T B-fragment, no transpose needed
#pragma unroll
        for (int i = 0; i < 16; ++i) {
            float p = __expf(sc[i] - m_new);
            psum += p;
            pf[i] = (_Float16)p;
        }
        l_run = l_run * corr + psum;
        m_run = m_new;
#pragma unroll
        for (int t = 0; t < 4; ++t)
#pragma unroll
            for (int r = 0; r < 8; ++r) acc[t][r] *= corr;

        // ---- GEMM2: O^T tiles += V^T x P^T ----
#pragma unroll
        for (int t = 0; t < 4; ++t) {
            const _Float16* Vrow = Vbase + (size_t)(t * 16 + n) * SD + kbase;
            v16h vf = load_frag_row(Vrow, grp * 8, 16 + grp * 8);
            acc[t] = __builtin_amdgcn_wmma_f32_16x16x32_f16(false, vf, false, pf,
                                                            (short)0, acc[t], false, false);
        }
    }

    // ---- epilogue: normalize + fused output transform cos(th)+sin(th) ----
    float l_tot = l_run + __shfl_xor(l_run, 16, 32);
    float inv_l = 1.0f / l_tot;
    float* orow = out + ((size_t)b * SD + (size_t)qg) * (HD * DH);
#pragma unroll
    for (int t = 0; t < 4; ++t) {
#pragma unroll
        for (int r = 0; r < 8; ++r) {
            int   f = t * 16 + r + 8 * grp;
            int   d = h * DH + f;
            float o = acc[t][r] * inv_l;
            float th = o / (1.0f + fabsf(w_out[d])) + b_out[d];
            float so, co; __sincosf(th, &so, &co);
            orow[d] = co + so;
        }
    }
}

extern "C" void kernel_launch(void* const* d_in, const int* in_sizes, int n_in,
                              void* d_out, int out_size, void* d_ws, size_t ws_size,
                              hipStream_t stream)
{
    const float* x     = (const float*)d_in[0];
    const float* w_q   = (const float*)d_in[1];
    const float* b_q   = (const float*)d_in[2];
    const float* phi_q = (const float*)d_in[3];
    const float* w_k   = (const float*)d_in[4];
    const float* b_k   = (const float*)d_in[5];
    const float* phi_k = (const float*)d_in[6];
    const float* w_v   = (const float*)d_in[7];
    const float* b_v   = (const float*)d_in[8];
    const float* w_out = (const float*)d_in[9];
    const float* b_out = (const float*)d_in[10];

    _Float16* Qb = (_Float16*)d_ws;                       // 16.78 MB
    _Float16* Kb = Qb + (size_t)BD * HD * SD * QK;        // 16.78 MB
    _Float16* Vt = Kb + (size_t)BD * HD * SD * QK;        //  8.39 MB

    const int total = BD * SD * HD * DH;                  // 4,194,304
    euler_qkv_kernel<<<(total + 255) / 256, 256, 0, stream>>>(
        x, w_q, b_q, phi_q, w_k, b_k, phi_k, w_v, b_v, Qb, Kb, Vt);

    const int waves = BD * HD * (SD / 16);                // 4096 waves
    euler_flash_attn_kernel<<<waves / 4, 128, 0, stream>>>(
        Qb, Kb, Vt, w_out, b_out, (float*)d_out);
}
